// AfmoeMoE_8186207666193
// MI455X (gfx1250) — compile-verified, matrix operand order checked by reading, not run on hardware
//
#include <hip/hip_runtime.h>

// ---------------- problem constants (from reference) ----------------
constexpr int Bc  = 2;
constexpr int Sc  = 512;
constexpr int Dc  = 1024;   // hidden
constexpr int Ec  = 8;      // experts
constexpr int Fc  = 768;    // routed intermediate
constexpr int FSc = 768;    // shared intermediate
constexpr int Tc  = Bc * Sc;      // 1024 tokens
constexpr int Nc  = Tc * 2;       // 2048 routed slots (top-2)

typedef __attribute__((ext_vector_type(16))) __bf16 v16bf;
typedef __attribute__((ext_vector_type(8)))  float  v8f;

#define BM 64
#define BN 64
#define BK 32
#define LSTR (BK + 8)   // 40 bf16 = 80B row stride: 16B-aligned, conflict-free

// ---- fragment loader: 16-bit 16x32 A-layout (B stored transposed -> same) ----
// lane l: row = l&15, half = l>>4
// VGPR j(0..3): K = 8*half + 2j, 2j+1 ; VGPR j(4..7): K = 16 + 8*half + 2(j-4)
__device__ inline v16bf frag_ld(const __bf16* __restrict__ tile, int row, int half) {
  const __bf16* p = tile + row * LSTR;
  v16bf v;
#pragma unroll
  for (int j = 0; j < 4; ++j) {
    int k = 8 * half + 2 * j;
    v[2 * j]     = p[k];
    v[2 * j + 1] = p[k + 1];
  }
#pragma unroll
  for (int j = 0; j < 4; ++j) {
    int k = 16 + 8 * half + 2 * j;
    v[8 + 2 * j]     = p[k];
    v[8 + 2 * j + 1] = p[k + 1];
  }
  return v;
}

// ---------------- small prep kernels ----------------
__global__ void k_cvt(const float* __restrict__ x, __bf16* __restrict__ xb, int n) {
  int i = (blockIdx.x * blockDim.x + threadIdx.x) * 4;
  if (i + 3 < n) {
    float4 v = *reinterpret_cast<const float4*>(x + i);
    xb[i + 0] = (__bf16)v.x; xb[i + 1] = (__bf16)v.y;
    xb[i + 2] = (__bf16)v.z; xb[i + 3] = (__bf16)v.w;
  }
}

__global__ void k_router(const float* __restrict__ x, const float* __restrict__ Wr,
                         float* __restrict__ scores) {
  int tid = blockIdx.x * blockDim.x + threadIdx.x;
  if (tid >= Tc * Ec) return;
  int t = tid >> 3, e = tid & 7;
  const float* xr = x + (size_t)t * Dc;
  float acc = 0.f;
  for (int k = 0; k < Dc; ++k) acc += xr[k] * Wr[k * Ec + e];
  scores[t * Ec + e] = 1.0f / (1.0f + __expf(-acc));   // sigmoid
}

__global__ void k_zero(int* __restrict__ cnt) {
  if (threadIdx.x < Ec) cnt[threadIdx.x] = 0;
}

__global__ void k_top2(const float* __restrict__ scores, const float* __restrict__ bias,
                       int* __restrict__ sel0, int* __restrict__ sel1,
                       float* __restrict__ w0, float* __restrict__ w1,
                       int* __restrict__ cnt) {
  int t = blockIdx.x * blockDim.x + threadIdx.x;
  if (t >= Tc) return;
  float s[Ec], sb[Ec];
#pragma unroll
  for (int e = 0; e < Ec; ++e) { s[e] = scores[t * Ec + e]; sb[e] = s[e] + bias[e]; }
  int i0 = 0;
#pragma unroll
  for (int e = 1; e < Ec; ++e) if (sb[e] > sb[i0]) i0 = e;   // strict > : lowest idx wins ties
  int i1 = (i0 == 0) ? 1 : 0;
#pragma unroll
  for (int e = 0; e < Ec; ++e) if (e != i0 && sb[e] > sb[i1]) i1 = e;
  float a = s[i0], b = s[i1];
  float d = a + b + 1e-20f;
  sel0[t] = i0; sel1[t] = i1;
  w0[t] = a / d; w1[t] = b / d;                              // ROUTE_SCALE = 1
  atomicAdd(&cnt[i0], 1);
  atomicAdd(&cnt[i1], 1);
}

__global__ void k_scan(const int* __restrict__ cnt, int* __restrict__ offs,
                       int* __restrict__ cur) {
  if (threadIdx.x == 0 && blockIdx.x == 0) {
    int acc = 0;
    for (int e = 0; e < Ec; ++e) { offs[e] = acc; cur[e] = acc; acc += cnt[e]; }
    offs[Ec] = acc;
  }
}

__global__ void k_fill(const int* __restrict__ sel0, const int* __restrict__ sel1,
                       const float* __restrict__ w0, const float* __restrict__ w1,
                       int* __restrict__ cur, int* __restrict__ tok,
                       float* __restrict__ wgt) {
  int t = blockIdx.x * blockDim.x + threadIdx.x;
  if (t >= Tc) return;
  int p0 = atomicAdd(&cur[sel0[t]], 1); tok[p0] = t; wgt[p0] = w0[t];
  int p1 = atomicAdd(&cur[sel1[t]], 1); tok[p1] = t; wgt[p1] = w1[t];
}

// ---------------- fused gate+up GEMM: H = silu(X@Wg) * (X@Wu) ----------------
// X gathered bf16 [*, Dk]; Wg/Wu f32 [E][Dk][Fn]; H bf16 [rows][Fn]
__global__ __launch_bounds__(256, 2)
void k_up(const __bf16* __restrict__ Xb, const float* __restrict__ Wg,
          const float* __restrict__ Wu, __bf16* __restrict__ H,
          const int* __restrict__ tokidx, const int* __restrict__ offs,
          int Dk, int Fn, int Ttot) {
  int e = blockIdx.z;
  int base = 0, cnt = Ttot;
  if (offs) { base = offs[e]; cnt = offs[e + 1] - base; }
  int row0 = blockIdx.y * BM;
  if (row0 >= cnt) return;
  int col0 = blockIdx.x * BN;

  const float* wg = Wg + (size_t)e * Dk * Fn;
  const float* wu = Wu + (size_t)e * Dk * Fn;

  __shared__ __align__(16) __bf16 sA [BM * LSTR];
  __shared__ __align__(16) __bf16 sBg[BN * LSTR];
  __shared__ __align__(16) __bf16 sBu[BN * LSTR];

  int tid = threadIdx.x;
  // A loader: 64 rows x 32 cols, 8 bf16 (16B) per thread
  int rA = tid >> 2, cA = (tid & 3) * 8;
  int lrowA = row0 + rA;
  int tokA = offs ? tokidx[base + (lrowA < cnt ? lrowA : 0)] : lrowA;
  const __bf16* aSrc = Xb + (size_t)tokA * Dk + cA;
  // B loader: 32 k-rows x 64 cols, 8 f32 per thread per matrix, transposed store
  int rB = tid >> 3, cB = (tid & 7) * 8;

  int lane = tid & 31, wave = tid >> 5;
  int wm = wave & 3, wn = wave >> 2;
  int m = lane & 15, hl = lane >> 4;

  v8f cg0 = {}, cg1 = {}, cu0 = {}, cu1 = {};

  for (int k0 = 0; k0 < Dk; k0 += BK) {
    *reinterpret_cast<uint4*>(&sA[rA * LSTR + cA]) =
        *reinterpret_cast<const uint4*>(aSrc + k0);
    {
      const float* pg = wg + (size_t)(k0 + rB) * Fn + col0 + cB;
      const float* pu = wu + (size_t)(k0 + rB) * Fn + col0 + cB;
      float4 g0 = reinterpret_cast<const float4*>(pg)[0];
      float4 g1 = reinterpret_cast<const float4*>(pg)[1];
      float4 u0 = reinterpret_cast<const float4*>(pu)[0];
      float4 u1 = reinterpret_cast<const float4*>(pu)[1];
      if (k0 + BK < Dk) {
        __builtin_prefetch(pg + (size_t)BK * Fn, 0, 1);
        __builtin_prefetch(pu + (size_t)BK * Fn, 0, 1);
      }
      sBg[(cB + 0) * LSTR + rB] = (__bf16)g0.x; sBg[(cB + 1) * LSTR + rB] = (__bf16)g0.y;
      sBg[(cB + 2) * LSTR + rB] = (__bf16)g0.z; sBg[(cB + 3) * LSTR + rB] = (__bf16)g0.w;
      sBg[(cB + 4) * LSTR + rB] = (__bf16)g1.x; sBg[(cB + 5) * LSTR + rB] = (__bf16)g1.y;
      sBg[(cB + 6) * LSTR + rB] = (__bf16)g1.z; sBg[(cB + 7) * LSTR + rB] = (__bf16)g1.w;
      sBu[(cB + 0) * LSTR + rB] = (__bf16)u0.x; sBu[(cB + 1) * LSTR + rB] = (__bf16)u0.y;
      sBu[(cB + 2) * LSTR + rB] = (__bf16)u0.z; sBu[(cB + 3) * LSTR + rB] = (__bf16)u0.w;
      sBu[(cB + 4) * LSTR + rB] = (__bf16)u1.x; sBu[(cB + 5) * LSTR + rB] = (__bf16)u1.y;
      sBu[(cB + 6) * LSTR + rB] = (__bf16)u1.z; sBu[(cB + 7) * LSTR + rB] = (__bf16)u1.w;
    }
    __syncthreads();
    v16bf a  = frag_ld(&sA [(wm * 16) * LSTR], m, hl);
    v16bf g0 = frag_ld(&sBg[(wn * 32) * LSTR], m, hl);
    v16bf g1 = frag_ld(&sBg[(wn * 32 + 16) * LSTR], m, hl);
    v16bf u0 = frag_ld(&sBu[(wn * 32) * LSTR], m, hl);
    v16bf u1 = frag_ld(&sBu[(wn * 32 + 16) * LSTR], m, hl);
    cg0 = __builtin_amdgcn_wmma_f32_16x16x32_bf16(false, a, false, g0, (short)0, cg0, false, false);
    cg1 = __builtin_amdgcn_wmma_f32_16x16x32_bf16(false, a, false, g1, (short)0, cg1, false, false);
    cu0 = __builtin_amdgcn_wmma_f32_16x16x32_bf16(false, a, false, u0, (short)0, cu0, false, false);
    cu1 = __builtin_amdgcn_wmma_f32_16x16x32_bf16(false, a, false, u1, (short)0, cu1, false, false);
    __syncthreads();
  }
  // epilogue: silu(g)*u -> bf16 H   (C layout: M = r + 8*half, N = lane&15)
#pragma unroll
  for (int sub = 0; sub < 2; ++sub) {
    v8f g = sub ? cg1 : cg0;
    v8f u = sub ? cu1 : cu0;
    int col = col0 + wn * 32 + sub * 16 + m;
#pragma unroll
    for (int r = 0; r < 8; ++r) {
      int lrow = row0 + wm * 16 + r + 8 * hl;
      if (lrow < cnt) {
        float gv = g[r];
        float hv = gv / (1.0f + __expf(-gv)) * u[r];
        H[(size_t)(base + lrow) * Fn + col] = (__bf16)hv;
      }
    }
  }
}

// ---------------- down GEMM: Out (+)= (H @ Wd) * w ----------------
__global__ __launch_bounds__(256, 2)
void k_down(const __bf16* __restrict__ A, const float* __restrict__ W,
            float* __restrict__ Out, const int* __restrict__ tokidx,
            const float* __restrict__ wgt, const int* __restrict__ offs,
            int Kd, int Nn, int Ttot) {
  int e = blockIdx.z;
  int base = 0, cnt = Ttot;
  if (offs) { base = offs[e]; cnt = offs[e + 1] - base; }
  int row0 = blockIdx.y * BM;
  if (row0 >= cnt) return;
  int col0 = blockIdx.x * BN;

  const float* wd = W + (size_t)e * Kd * Nn;

  __shared__ __align__(16) __bf16 sA[BM * LSTR];
  __shared__ __align__(16) __bf16 sB[BN * LSTR];

  int tid = threadIdx.x;
  int rA = tid >> 2, cA = (tid & 3) * 8;
  int lrowA = row0 + rA;
  const __bf16* aSrc = A + (size_t)(base + (lrowA < cnt ? lrowA : 0)) * Kd + cA;
  int rB = tid >> 3, cB = (tid & 7) * 8;

  int lane = tid & 31, wave = tid >> 5;
  int wm = wave & 3, wn = wave >> 2;
  int m = lane & 15, hl = lane >> 4;

  v8f c0 = {}, c1 = {};

  for (int k0 = 0; k0 < Kd; k0 += BK) {
    *reinterpret_cast<uint4*>(&sA[rA * LSTR + cA]) =
        *reinterpret_cast<const uint4*>(aSrc + k0);
    {
      const float* pw = wd + (size_t)(k0 + rB) * Nn + col0 + cB;
      float4 b0 = reinterpret_cast<const float4*>(pw)[0];
      float4 b1 = reinterpret_cast<const float4*>(pw)[1];
      if (k0 + BK < Kd) __builtin_prefetch(pw + (size_t)BK * Nn, 0, 1);
      sB[(cB + 0) * LSTR + rB] = (__bf16)b0.x; sB[(cB + 1) * LSTR + rB] = (__bf16)b0.y;
      sB[(cB + 2) * LSTR + rB] = (__bf16)b0.z; sB[(cB + 3) * LSTR + rB] = (__bf16)b0.w;
      sB[(cB + 4) * LSTR + rB] = (__bf16)b1.x; sB[(cB + 5) * LSTR + rB] = (__bf16)b1.y;
      sB[(cB + 6) * LSTR + rB] = (__bf16)b1.z; sB[(cB + 7) * LSTR + rB] = (__bf16)b1.w;
    }
    __syncthreads();
    v16bf a  = frag_ld(&sA[(wm * 16) * LSTR], m, hl);
    v16bf b0 = frag_ld(&sB[(wn * 32) * LSTR], m, hl);
    v16bf b1 = frag_ld(&sB[(wn * 32 + 16) * LSTR], m, hl);
    c0 = __builtin_amdgcn_wmma_f32_16x16x32_bf16(false, a, false, b0, (short)0, c0, false, false);
    c1 = __builtin_amdgcn_wmma_f32_16x16x32_bf16(false, a, false, b1, (short)0, c1, false, false);
    __syncthreads();
  }
#pragma unroll
  for (int sub = 0; sub < 2; ++sub) {
    v8f c = sub ? c1 : c0;
    int col = col0 + wn * 32 + sub * 16 + m;
#pragma unroll
    for (int r = 0; r < 8; ++r) {
      int lrow = row0 + wm * 16 + r + 8 * hl;
      if (lrow < cnt) {
        if (offs) {
          int t = tokidx[base + lrow];
          float sc = wgt[base + lrow];
          unsafeAtomicAdd(&Out[(size_t)t * Nn + col], c[r] * sc);  // global_atomic_add_f32
        } else {
          Out[(size_t)lrow * Nn + col] = c[r];   // shared expert initializes output
        }
      }
    }
  }
}

// ---------------- launcher ----------------
extern "C" void kernel_launch(void* const* d_in, const int* in_sizes, int n_in,
                              void* d_out, int out_size, void* d_ws, size_t ws_size,
                              hipStream_t stream) {
  const float* x    = (const float*)d_in[0];
  const float* Wr   = (const float*)d_in[1];
  const float* bias = (const float*)d_in[2];
  const float* Wg   = (const float*)d_in[3];
  const float* Wu   = (const float*)d_in[4];
  const float* Wd   = (const float*)d_in[5];
  const float* Wgs  = (const float*)d_in[6];
  const float* Wus  = (const float*)d_in[7];
  const float* Wds  = (const float*)d_in[8];
  float* out = (float*)d_out;

  char* ws = (char*)d_ws;
  size_t off = 0;
  auto alloc = [&](size_t bytes) -> char* {
    char* p = ws + off;
    off += (bytes + 255) & ~(size_t)255;
    return p;
  };
  __bf16* XB  = (__bf16*)alloc((size_t)Tc * Dc * 2);
  __bf16* HS  = (__bf16*)alloc((size_t)Tc * FSc * 2);
  __bf16* HR  = (__bf16*)alloc((size_t)Nc * Fc * 2);
  float*  SCO = (float*) alloc((size_t)Tc * Ec * 4);
  int*    SEL0= (int*)   alloc((size_t)Tc * 4);
  int*    SEL1= (int*)   alloc((size_t)Tc * 4);
  float*  W0  = (float*) alloc((size_t)Tc * 4);
  float*  W1  = (float*) alloc((size_t)Tc * 4);
  int*    CNT = (int*)   alloc(Ec * 4);
  int*    OFFS= (int*)   alloc((Ec + 1) * 4);
  int*    CUR = (int*)   alloc(Ec * 4);
  int*    TOK = (int*)   alloc((size_t)Nc * 4);
  float*  WGT = (float*) alloc((size_t)Nc * 4);
  (void)ws_size; (void)in_sizes; (void)n_in; (void)out_size;

  k_cvt   <<<(Tc * Dc / 4 + 255) / 256, 256, 0, stream>>>(x, XB, Tc * Dc);
  k_router<<<(Tc * Ec + 255) / 256,    256, 0, stream>>>(x, Wr, SCO);
  k_zero  <<<1, 32, 0, stream>>>(CNT);
  k_top2  <<<(Tc + 255) / 256, 256, 0, stream>>>(SCO, bias, SEL0, SEL1, W0, W1, CNT);
  k_scan  <<<1, 1, 0, stream>>>(CNT, OFFS, CUR);
  k_fill  <<<(Tc + 255) / 256, 256, 0, stream>>>(SEL0, SEL1, W0, W1, CUR, TOK, WGT);

  // shared expert up: Hs = silu(x@Wg_s) * (x@Wu_s)
  k_up<<<dim3(FSc / BN, Tc / BM, 1), 256, 0, stream>>>(
      XB, Wgs, Wus, HS, nullptr, nullptr, Dc, FSc, Tc);
  // routed experts up (grouped GEMM over compacted lists)
  k_up<<<dim3(Fc / BN, Nc / BM, Ec), 256, 0, stream>>>(
      XB, Wg, Wu, HR, TOK, OFFS, Dc, Fc, Tc);
  // shared expert down: plain store (initializes d_out)
  k_down<<<dim3(Dc / BN, Tc / BM, 1), 256, 0, stream>>>(
      HS, Wds, out, nullptr, nullptr, nullptr, FSc, Dc, Tc);
  // routed experts down: scaled atomic scatter-add
  k_down<<<dim3(Dc / BN, Nc / BM, Ec), 256, 0, stream>>>(
      HR, Wd, out, TOK, WGT, OFFS, Fc, Dc, Tc);
}
